// Decoder_9758165696839
// MI455X (gfx1250) — matrix operand order repaired and checked by reading, not compiled, not run on hardware
//
#include <hip/hip_runtime.h>

// ---------------- problem constants ----------------
#define B_  32
#define T_  64
#define S_  64
#define E_  512
#define H_  1024
#define V_  32000
#define TM1 63            // T-1 steps
#define MROWS (B_*TM1)    // 2016 generator rows
#define MPAD  2048        // padded generator rows
#define KSPLIT 4          // split-K factor for recurrent GEMMs

// ---------------- WMMA types ----------------
typedef __bf16 bf16x16 __attribute__((ext_vector_type(16)));
typedef float  f32x8   __attribute__((ext_vector_type(8)));
typedef int    gv4i    __attribute__((vector_size(16)));   // matches builtin param pointee

union Frag16 {
  bf16x16 v;
  unsigned int u[8];
};

struct __align__(16) BfVec8 { unsigned int u[4]; };  // 8 bf16 = 16 bytes

// CDNA5 async global->LDS path (guarded; falls back to sync staging)
#if __has_builtin(__builtin_amdgcn_global_load_async_to_lds_b128) && \
    __has_builtin(__builtin_amdgcn_s_wait_asynccnt)
#define HAVE_ASYNC_LDS 1
#else
#define HAVE_ASYNC_LDS 0
#endif

__device__ __forceinline__ void ldg16_to_lds(const unsigned short* g, unsigned short* l) {
#if HAVE_ASYNC_LDS
  __builtin_amdgcn_global_load_async_to_lds_b128(
      (__attribute__((address_space(1))) gv4i*)(void*)g,
      (__attribute__((address_space(3))) gv4i*)(void*)l, 0, 0);
#else
  *(BfVec8*)l = *(const BfVec8*)g;
#endif
}

__device__ __forceinline__ void stage_fence() {
#if HAVE_ASYNC_LDS
  __builtin_amdgcn_s_wait_asynccnt(0);
#endif
  __syncthreads();
}

__device__ __forceinline__ unsigned short f2bf(float f) {
  unsigned int u = __float_as_uint(f);
  u = u + 0x7fffu + ((u >> 16) & 1u);   // round to nearest even
  return (unsigned short)(u >> 16);
}

__device__ __forceinline__ float sigmoidf_(float x) {
  return 1.0f / (1.0f + __expf(-x));
}

// ---------------- one-time prep kernels ----------------

// WT[rowoff+k][n] = bf16(W[n][k]);  W is [N][K] row-major, WT row stride ldwt
__global__ __launch_bounds__(256) void transpose_cast_kernel(
    const float* __restrict__ W, unsigned short* __restrict__ WT,
    int N, int K, int ldwt, int rowoff) {
  size_t idx = (size_t)blockIdx.x * 256 + threadIdx.x;
  size_t total = (size_t)N * K;
  if (idx >= total) return;
  int k = (int)(idx / N);
  int n = (int)(idx % N);
  WT[(size_t)(rowoff + k) * ldwt + n] = f2bf(W[(size_t)n * K + k]);
}

__global__ __launch_bounds__(256) void bias_sum_kernel(
    const float* __restrict__ a, const float* __restrict__ b,
    float* __restrict__ o, int n) {
  int i = blockIdx.x * 256 + threadIdx.x;
  if (i < n) o[i] = a[i] + b[i];
}

__global__ __launch_bounds__(256) void zero_sh_kernel(unsigned short* __restrict__ p, size_t n) {
  size_t i = (size_t)blockIdx.x * 256 + threadIdx.x;
  if (i < n) p[i] = 0;
}

// c_ws = c0; scatter bf16(h0) into the recurrent concat buffers
__global__ __launch_bounds__(256) void init_state_kernel(
    const float* __restrict__ h0, const float* __restrict__ c0,
    float* __restrict__ c_ws,
    unsigned short* __restrict__ inp0, unsigned short* __restrict__ inp1) {
  int idx = blockIdx.x * 256 + threadIdx.x;       // 2*32*1024
  if (idx >= 2 * B_ * H_) return;
  c_ws[idx] = c0[idx];
  int l = idx >> 15;
  int b = (idx >> 10) & (B_ - 1);
  int j = idx & (H_ - 1);
  unsigned short hb = f2bf(h0[idx]);
  if (l == 0) inp0[b * 2560 + 1536 + j] = hb;     // layer0 h_prev slot
  else        inp1[b * 2048 + 1024 + j] = hb;     // layer1 h_prev slot
}

// inp0[b][0:512) = bf16(emb[x[b][t]])
__global__ __launch_bounds__(256) void embed_slice_kernel(
    const int* __restrict__ x, const float* __restrict__ emb,
    unsigned short* __restrict__ inp0, int t) {
  int idx = blockIdx.x * 256 + threadIdx.x;       // 32*512
  if (idx >= B_ * E_) return;
  int b = idx >> 9, e = idx & (E_ - 1);
  int tok = x[b * T_ + t];
  inp0[b * 2560 + e] = f2bf(emb[(size_t)tok * E_ + e]);
}

// ---------------- small GEMM, split-K ----------------
// Cpart[p][M x N] = A[M x Kc](bf16) * BT[Kc x N](bf16) for K-chunk p = blockIdx.y.
// One 16x16 output tile per wave; pure pointer-increment addressing.
__global__ __launch_bounds__(256) void gemm_small_kernel(
    const unsigned short* __restrict__ A, int lda,
    const unsigned short* __restrict__ BT, int ldb,
    float* __restrict__ Cpart, int ldc,
    int M, int N, int K) {
  const int wave = (int)((blockIdx.x * 256 + threadIdx.x) >> 5);
  const int lane = threadIdx.x & 31;
  const int MT = M >> 4;
  const int NT = N >> 4;
  if (wave >= MT * NT) return;
  const int mt = wave % MT;
  const int nt = wave / MT;
  const int m0 = mt << 4, n0 = nt << 4;
  const int p = blockIdx.y;
  const int Kc = K / KSPLIT;
  const int kstart = p * Kc;

  f32x8 acc = {0.f, 0.f, 0.f, 0.f, 0.f, 0.f, 0.f, 0.f};
  const int kb = (lane >> 4) << 3;                   // 0 or 8: A K-half select
  const unsigned short* Ap = A + (size_t)(m0 + (lane & 15)) * lda + kstart + kb;
  const unsigned short* Bp = BT + (size_t)(kstart + kb) * ldb + n0 + (lane & 15);
  const size_t bstep = (size_t)32 * ldb;

  for (int k0 = 0; k0 < Kc; k0 += 32) {
    Frag16 af, bf;
    // A fragment: 16-bit A layout (16x32): VGPR i -> K pair
#pragma unroll
    for (int i = 0; i < 8; ++i) {
      int k = (i < 4) ? (i << 1) : (16 + ((i - 4) << 1));
      af.u[i] = *(const unsigned int*)(Ap + k);
    }
    // B fragment: lane = N column, VGPR i -> K pair (same order as A)
#pragma unroll
    for (int i = 0; i < 8; ++i) {
      int k = (i < 4) ? (i << 1) : (16 + ((i - 4) << 1));
      bf.u[i] = (unsigned int)Bp[(size_t)k * ldb] | ((unsigned int)Bp[(size_t)(k + 1) * ldb] << 16);
    }

    __builtin_prefetch(Bp + bstep, 0, 1);

    acc = __builtin_amdgcn_wmma_f32_16x16x32_bf16(
        false, af.v, false, bf.v, (short)0, acc, false, false);
    Ap += 32;
    Bp += bstep;
  }

  const int n = n0 + (lane & 15);
  const int mb = m0 + ((lane >> 4) << 3);
  float* Cp = Cpart + (size_t)p * M * ldc;
#pragma unroll
  for (int v = 0; v < 8; ++v)
    Cp[(size_t)(mb + v) * ldc + n] = acc[v];
}

// ---------------- LSTM elementwise cell ----------------
// g: [KSPLIT][32][4096] partial pre-activation gates; bsum = bih+bhh.
__global__ __launch_bounds__(256) void lstm_cell_kernel(
    const float* __restrict__ g, const float* __restrict__ bsum,
    float* __restrict__ c,
    unsigned short* __restrict__ d1, int s1, int o1,
    unsigned short* __restrict__ d2, int s2, int o2) {
  int idx = blockIdx.x * 256 + threadIdx.x;        // 32*1024
  if (idx >= B_ * H_) return;
  int b = idx >> 10, j = idx & (H_ - 1);
  float i_ = bsum[j];
  float f_ = bsum[H_ + j];
  float g_ = bsum[2 * H_ + j];
  float o_ = bsum[3 * H_ + j];
#pragma unroll
  for (int p = 0; p < KSPLIT; ++p) {
    const float* gb = g + ((size_t)p * B_ + b) * 4 * H_;
    i_ += gb[j];
    f_ += gb[H_ + j];
    g_ += gb[2 * H_ + j];
    o_ += gb[3 * H_ + j];
  }
  float cn = sigmoidf_(f_) * c[idx] + sigmoidf_(i_) * tanhf(g_);
  float hn = sigmoidf_(o_) * tanhf(cn);
  c[idx] = cn;
  unsigned short hb = f2bf(hn);
  d1[b * s1 + o1 + j] = hb;
  d2[b * s2 + o2 + j] = hb;
}

// ---------------- attention: scores + softmax + context ----------------
// zap: [KSPLIT][32][1024] partials of z@Wa^T; ba bias. Writes ctx bf16 -> outcat[b][1024+h].
__global__ __launch_bounds__(256) void attn_kernel(
    const float* __restrict__ zap, const float* __restrict__ ba,
    const float* __restrict__ x_enc, unsigned short* __restrict__ outcat) {
  __shared__ float zsh[H_];
  __shared__ float red[256];
  __shared__ float wsm[S_];
  const int b = blockIdx.x;
  const int tid = threadIdx.x;
  const float* xb = x_enc + (size_t)b * S_ * H_;
  // reduce split-K partials + bias into shared z vector
  for (int h = tid; h < H_; h += 256) {
    float v = ba[h];
#pragma unroll
    for (int p = 0; p < KSPLIT; ++p)
      v += zap[((size_t)p * B_ + b) * H_ + h];
    zsh[h] = v;
  }
  __syncthreads();
  // scores
  const int s = tid >> 2;
  const int part = tid & 3;
  float sum = 0.f;
  {
    const float* xr = xb + s * H_ + part * 256;
    const float* zr = zsh + part * 256;
    for (int i = 0; i < 256; ++i) sum += zr[i] * xr[i];
  }
  red[tid] = sum;
  __syncthreads();
  if (part == 0) wsm[s] = red[tid] + red[tid + 1] + red[tid + 2] + red[tid + 3];
  __syncthreads();
  if (tid == 0) {
    float mx = wsm[0];
    for (int i = 1; i < S_; ++i) mx = fmaxf(mx, wsm[i]);
    float ssum = 0.f;
    for (int i = 0; i < S_; ++i) { float e = __expf(wsm[i] - mx); wsm[i] = e; ssum += e; }
    float inv = 1.f / ssum;
    for (int i = 0; i < S_; ++i) wsm[i] *= inv;
  }
  __syncthreads();
  for (int h = tid; h < H_; h += 256) {
    float acc = 0.f;
    for (int s2 = 0; s2 < S_; ++s2) acc += wsm[s2] * xb[s2 * H_ + h];
    outcat[b * 2048 + 1024 + h] = f2bf(acc);
  }
}

// ---------------- tanh epilogue of output linear ----------------
// op: [KSPLIT][32][1024] partials of outcat@Wl^T; bl bias.
__global__ __launch_bounds__(256) void out_epilogue_kernel(
    const float* __restrict__ op, const float* __restrict__ bl,
    unsigned short* __restrict__ inp0,
    unsigned short* __restrict__ outs_bf, int t) {
  int idx = blockIdx.x * 256 + threadIdx.x;        // 32*1024
  if (idx >= B_ * H_) return;
  int b = idx >> 10, j = idx & (H_ - 1);
  float v = bl[j];
#pragma unroll
  for (int p = 0; p < KSPLIT; ++p)
    v += op[((size_t)p * B_ + b) * H_ + j];
  unsigned short hb = f2bf(tanhf(v));
  inp0[b * 2560 + 512 + j] = hb;                          // prev for next step
  outs_bf[((size_t)b * TM1 + t) * H_ + j] = hb;           // generator A row b*63+t
}

// ---------------- generator GEMM: [2016 x 32000] = A[2048 x 1024] * BT[1024 x 32000] + bg -----
// 256 threads = 8 waves; 64(M) x 128(N) block tile; each wave 2x2 WMMA tiles; K step 32.
// LDS double-buffered; async global->LDS when the toolchain exposes the gfx1250 builtins.
__global__ __launch_bounds__(256) void gemm_gen_kernel(
    const unsigned short* __restrict__ A,   // [MPAD][1024], pad rows zeroed
    const unsigned short* __restrict__ BT,  // [1024][32000]
    const float* __restrict__ bias,         // [32000]
    float* __restrict__ C,                  // [2016][32000]
    int M, int N, int K) {
  __shared__ __align__(16) unsigned short As[2][64][32];
  __shared__ __align__(16) unsigned short Bs[2][32][128];

  const int nb = N / 128;
  const int bm = blockIdx.x / nb;
  const int bn = blockIdx.x % nb;
  const int m0 = bm * 64, n0 = bn * 128;
  const int tid = threadIdx.x;
  const int wave = tid >> 5, lane = tid & 31;
  const int wm = wave >> 2, wn = wave & 3;

  // global->LDS staging coords
  const int ar = tid >> 2;                  // 0..63
  const int ac = (tid & 3) << 3;            // 0,8,16,24
  const int br = tid >> 3;                  // 0..31
  const int bc = (tid & 7) << 4;            // 0..112

  const unsigned short* Ag = A + (size_t)(m0 + ar) * (size_t)K + ac;
  const unsigned short* Bg = BT + (size_t)br * (size_t)N + (n0 + bc);
  const size_t bstep = (size_t)32 * (size_t)N;

  f32x8 zero8 = {0.f, 0.f, 0.f, 0.f, 0.f, 0.f, 0.f, 0.f};
  f32x8 acc[2][2] = {{zero8, zero8}, {zero8, zero8}};

  // preload tile 0
  ldg16_to_lds(Ag, &As[0][ar][ac]);
  ldg16_to_lds(Bg, &Bs[0][br][bc]);
  ldg16_to_lds(Bg + 8, &Bs[0][br][bc + 8]);
  stage_fence();

  const unsigned short* Agn = Ag + 32;      // next-tile staging pointers
  const unsigned short* Bgn = Bg + bstep;

  const int KT = K >> 5;
  for (int kt = 0; kt < KT; ++kt) {
    const int cur = kt & 1, nxt = cur ^ 1;
    if (kt + 1 < KT) {
      ldg16_to_lds(Agn, &As[nxt][ar][ac]);
      ldg16_to_lds(Bgn, &Bs[nxt][br][bc]);
      ldg16_to_lds(Bgn + 8, &Bs[nxt][br][bc + 8]);
      Agn += 32;
      Bgn += bstep;
    }

    Frag16 af[2], bfr[2];
    const int kb = (lane >> 4) << 3;
#pragma unroll
    for (int mi = 0; mi < 2; ++mi) {
      const unsigned short* Ar = &As[cur][wm * 32 + mi * 16 + (lane & 15)][0];
#pragma unroll
      for (int i = 0; i < 8; ++i) {
        int k = kb + ((i < 4) ? (i << 1) : (16 + ((i - 4) << 1)));
        af[mi].u[i] = *(const unsigned int*)(Ar + k);
      }
    }
#pragma unroll
    for (int ni = 0; ni < 2; ++ni) {
      const unsigned short* Br = &Bs[cur][kb][wn * 32 + ni * 16 + (lane & 15)];
#pragma unroll
      for (int i = 0; i < 8; ++i) {
        int k = (i < 4) ? (i << 1) : (16 + ((i - 4) << 1));
        bfr[ni].u[i] = (unsigned int)Br[k * 128] | ((unsigned int)Br[(k + 1) * 128] << 16);
      }
    }
#pragma unroll
    for (int mi = 0; mi < 2; ++mi)
#pragma unroll
      for (int ni = 0; ni < 2; ++ni)
        acc[mi][ni] = __builtin_amdgcn_wmma_f32_16x16x32_bf16(
            false, af[mi].v, false, bfr[ni].v, (short)0, acc[mi][ni], false, false);
    stage_fence();
  }

  // epilogue: add bias, guarded f32 stores
#pragma unroll
  for (int mi = 0; mi < 2; ++mi) {
#pragma unroll
    for (int ni = 0; ni < 2; ++ni) {
      const int n = n0 + wn * 32 + ni * 16 + (lane & 15);
      const int mbase = m0 + wm * 32 + mi * 16 + ((lane >> 4) << 3);
      const float bn = bias[n];
#pragma unroll
      for (int v = 0; v < 8; ++v) {
        int m = mbase + v;
        if (m < M) C[(size_t)m * (size_t)N + n] = acc[mi][ni][v] + bn;
      }
    }
  }
}

// ---------------- in-place log-softmax per row ----------------
__global__ __launch_bounds__(256) void log_softmax_kernel(float* __restrict__ C, int N) {
  __shared__ float red[256];
  float* r = C + (size_t)blockIdx.x * (size_t)N;
  const int tid = threadIdx.x;
  float mx = -3.402823466e38f;
  for (int i = tid; i < N; i += 256) mx = fmaxf(mx, r[i]);
  red[tid] = mx;
  __syncthreads();
  for (int s = 128; s > 0; s >>= 1) {
    if (tid < s) red[tid] = fmaxf(red[tid], red[tid + s]);
    __syncthreads();
  }
  mx = red[0];
  __syncthreads();
  float sum = 0.f;
  for (int i = tid; i < N; i += 256) sum += __expf(r[i] - mx);
  red[tid] = sum;
  __syncthreads();
  for (int s = 128; s > 0; s >>= 1) {
    if (tid < s) red[tid] += red[tid + s];
    __syncthreads();
  }
  const float lse = mx + __logf(red[0]);
  for (int i = tid; i < N; i += 256) r[i] = r[i] - lse;
}

// ---------------- host driver ----------------
#define CDIV(a, b) (((a) + (b) - 1) / (b))

extern "C" void kernel_launch(void* const* d_in, const int* in_sizes, int n_in,
                              void* d_out, int out_size, void* d_ws, size_t ws_size,
                              hipStream_t stream) {
  (void)in_sizes; (void)n_in; (void)out_size; (void)ws_size;
  const int*   x     = (const int*)  d_in[0];
  const float* h0    = (const float*)d_in[1];
  const float* c0    = (const float*)d_in[2];
  const float* x_enc = (const float*)d_in[3];
  const float* emb   = (const float*)d_in[4];
  const float* Wih0  = (const float*)d_in[5];
  const float* Whh0  = (const float*)d_in[6];
  const float* bih0  = (const float*)d_in[7];
  const float* bhh0  = (const float*)d_in[8];
  const float* Wih1  = (const float*)d_in[9];
  const float* Whh1  = (const float*)d_in[10];
  const float* bih1  = (const float*)d_in[11];
  const float* bhh1  = (const float*)d_in[12];
  const float* Wa    = (const float*)d_in[13];
  const float* ba    = (const float*)d_in[14];
  const float* Wl    = (const float*)d_in[15];
  const float* bl    = (const float*)d_in[16];
  const float* Wg    = (const float*)d_in[17];
  const float* bg    = (const float*)d_in[18];
  float* out = (float*)d_out;

  // workspace carve-out
  char* base = (char*)d_ws;
  size_t off = 0;
  auto alloc = [&](size_t bytes) -> void* {
    void* p = base + off;
    off += (bytes + 255) & ~(size_t)255;
    return p;
  };
  unsigned short* WT0     = (unsigned short*)alloc((size_t)2560 * 4096 * 2);  // [Wih0|Whh0]^T
  unsigned short* WT1     = (unsigned short*)alloc((size_t)2048 * 4096 * 2);  // [Wih1|Whh1]^T
  unsigned short* WaT     = (unsigned short*)alloc((size_t)1024 * 1024 * 2);
  unsigned short* WlT     = (unsigned short*)alloc((size_t)2048 * 1024 * 2);
  unsigned short* WgT     = (unsigned short*)alloc((size_t)1024 * V_ * 2);
  unsigned short* inp0    = (unsigned short*)alloc((size_t)B_ * 2560 * 2);    // [e | prev | h0_prev]
  unsigned short* inp1    = (unsigned short*)alloc((size_t)B_ * 2048 * 2);    // [h0_new | h1_prev]
  unsigned short* outcat  = (unsigned short*)alloc((size_t)B_ * 2048 * 2);    // [z | ctx]
  unsigned short* outs_bf = (unsigned short*)alloc((size_t)MPAD * H_ * 2);    // generator A (padded)
  float* g     = (float*)alloc((size_t)KSPLIT * B_ * 4 * H_ * 4);             // gate partials
  float* za    = (float*)alloc((size_t)KSPLIT * B_ * H_ * 4);                 // attn-proj partials
  float* oatt  = (float*)alloc((size_t)KSPLIT * B_ * H_ * 4);                 // out-linear partials
  float* c_ws  = (float*)alloc((size_t)2 * B_ * H_ * 4);
  float* bsum0 = (float*)alloc((size_t)4 * H_ * 4);
  float* bsum1 = (float*)alloc((size_t)4 * H_ * 4);

  // ---- one-time prep ----
  transpose_cast_kernel<<<CDIV((size_t)4096 * 1536, 256), 256, 0, stream>>>(Wih0, WT0, 4096, 1536, 4096, 0);
  transpose_cast_kernel<<<CDIV((size_t)4096 * 1024, 256), 256, 0, stream>>>(Whh0, WT0, 4096, 1024, 4096, 1536);
  transpose_cast_kernel<<<CDIV((size_t)4096 * 1024, 256), 256, 0, stream>>>(Wih1, WT1, 4096, 1024, 4096, 0);
  transpose_cast_kernel<<<CDIV((size_t)4096 * 1024, 256), 256, 0, stream>>>(Whh1, WT1, 4096, 1024, 4096, 1024);
  transpose_cast_kernel<<<CDIV((size_t)1024 * 1024, 256), 256, 0, stream>>>(Wa, WaT, 1024, 1024, 1024, 0);
  transpose_cast_kernel<<<CDIV((size_t)1024 * 2048, 256), 256, 0, stream>>>(Wl, WlT, 1024, 2048, 1024, 0);
  transpose_cast_kernel<<<CDIV((size_t)V_ * 1024, 256), 256, 0, stream>>>(Wg, WgT, V_, 1024, V_, 0);
  bias_sum_kernel<<<CDIV(4 * H_, 256), 256, 0, stream>>>(bih0, bhh0, bsum0, 4 * H_);
  bias_sum_kernel<<<CDIV(4 * H_, 256), 256, 0, stream>>>(bih1, bhh1, bsum1, 4 * H_);
  zero_sh_kernel<<<CDIV((size_t)B_ * 2560, 256), 256, 0, stream>>>(inp0, (size_t)B_ * 2560);
  zero_sh_kernel<<<CDIV((size_t)B_ * 2048, 256), 256, 0, stream>>>(inp1, (size_t)B_ * 2048);
  zero_sh_kernel<<<CDIV((size_t)B_ * 2048, 256), 256, 0, stream>>>(outcat, (size_t)B_ * 2048);
  zero_sh_kernel<<<CDIV((size_t)MPAD * H_, 256), 256, 0, stream>>>(outs_bf, (size_t)MPAD * H_);
  init_state_kernel<<<CDIV(2 * B_ * H_, 256), 256, 0, stream>>>(h0, c0, c_ws, inp0, inp1);

  // ---- recurrent decode: 63 sequential steps ----
  const dim3 gGate(64, KSPLIT);   // 512 tiles/chunk for N=4096
  const dim3 gProj(16, KSPLIT);   // 128 tiles/chunk for N=1024
  for (int t = 0; t < TM1; ++t) {
    embed_slice_kernel<<<CDIV(B_ * E_, 256), 256, 0, stream>>>(x, emb, inp0, t);
    // layer 0 gates: [32 x 2560] * [2560 x 4096], split-K x4
    gemm_small_kernel<<<gGate, 256, 0, stream>>>(inp0, 2560, WT0, 4096, g, 4096, 32, 4096, 2560);
    lstm_cell_kernel<<<CDIV(B_ * H_, 256), 256, 0, stream>>>(g, bsum0, c_ws, inp0, 2560, 1536, inp1, 2048, 0);
    // layer 1 gates: [32 x 2048] * [2048 x 4096], split-K x4
    gemm_small_kernel<<<gGate, 256, 0, stream>>>(inp1, 2048, WT1, 4096, g, 4096, 32, 4096, 2048);
    lstm_cell_kernel<<<CDIV(B_ * H_, 256), 256, 0, stream>>>(g, bsum1, c_ws + B_ * H_, inp1, 2048, 1024, outcat, 2048, 0);
    // attention projection: za = z @ Wa^T (A = z slice of outcat, lda=2048), split-K x4
    gemm_small_kernel<<<gProj, 256, 0, stream>>>(outcat, 2048, WaT, 1024, za, 1024, 32, 1024, 1024);
    attn_kernel<<<B_, 256, 0, stream>>>(za, ba, x_enc, outcat);
    // output linear: [32 x 2048] * [2048 x 1024], split-K x4
    gemm_small_kernel<<<gProj, 256, 0, stream>>>(outcat, 2048, WlT, 1024, oatt, 1024, 32, 1024, 2048);
    out_epilogue_kernel<<<CDIV(B_ * H_, 256), 256, 0, stream>>>(oatt, bl, inp0, outs_bf, t);
  }

  // ---- generator GEMM + log-softmax ----
  gemm_gen_kernel<<<(MPAD / 64) * (V_ / 128), 256, 0, stream>>>(outs_bf, WgT, bg, out, MROWS, V_, 1024);
  log_softmax_kernel<<<MROWS, 256, 0, stream>>>(out, V_);
}
